// TriModalAttention_6270652252305
// MI455X (gfx1250) — matrix operand order repaired and checked by reading, not compiled, not run on hardware
//
#include <hip/hip_runtime.h>
#include <hip/hip_bf16.h>

// ---------------------------------------------------------------------------
// TriModalAttention for MI455X (gfx1250, wave32, WMMA)
//   B=16, S=512, F=1024
//   feat_br = tanh(x1 @ W1 + x2 @ W2 + b)      (single GEMM, K=2048, bf16 WMMA)
//   out_br  = (softmax(q @ feat^T) @ q) * feat (bf16 WMMA both GEMMs)
// Workspace: 3 * 16*512*1024 bf16 features = 48 MB in d_ws.
// CDNA5 paths: v_wmma_f32_16x16x32_bf16, global_load_async_to_lds_b128
//   (+ s_wait_asynccnt), v_tanh_f32, global_prefetch_b8.
// ---------------------------------------------------------------------------

#define BDIM 16
#define SDIM 512
#define FDIM 1024
#define ROWS (BDIM * SDIM)   // 8192

typedef __attribute__((ext_vector_type(16))) __bf16 v16bf;
typedef __attribute__((ext_vector_type(8)))  float  v8f;

__device__ __forceinline__ unsigned short f2bf_bits(float f) {
    unsigned int u = __float_as_uint(f);
    u += 0x7FFFu + ((u >> 16) & 1u);           // round-to-nearest-even
    return (unsigned short)(u >> 16);
}
__device__ __forceinline__ __bf16 tobf(float f) {
    unsigned short h = f2bf_bits(f);
    __bf16 r;
    __builtin_memcpy(&r, &h, 2);
    return r;
}
__device__ __forceinline__ float bf2f(unsigned short h) {
    return __uint_as_float(((unsigned int)h) << 16);
}

// gfx1250 hardware tanh (V_TANH_F32) instead of the ~25-op libm expansion.
__device__ __forceinline__ float tanh_fast(float x) {
#if __has_builtin(__builtin_amdgcn_tanhf)
    return __builtin_amdgcn_tanhf(x);
#else
    float y;
    asm("v_tanh_f32 %0, %1" : "=v"(y) : "v"(x));
    return y;
#endif
}

// ===========================================================================
// Kernel 1: branch features.  grid = (F/128, ROWS/128, 3), block = 256 (8 waves)
// ===========================================================================
__global__ __launch_bounds__(256)
void branch_feat_kernel(const float* __restrict__ v,
                        const float* __restrict__ t,
                        const float* __restrict__ a,
                        const float* __restrict__ Wtv, const float* __restrict__ btv,
                        const float* __restrict__ Wta, const float* __restrict__ bta,
                        const float* __restrict__ Wav, const float* __restrict__ bav,
                        unsigned short* __restrict__ featws)
{
    const int br = blockIdx.z;
    const float* x1   = (br == 0) ? t : (br == 1) ? t : a;
    const float* x2   = (br == 0) ? v : (br == 1) ? a : v;
    const float* W    = (br == 0) ? Wtv : (br == 1) ? Wta : Wav;
    const float* bias = (br == 0) ? btv : (br == 1) ? bta : bav;
    unsigned short* feat = featws + (size_t)br * ROWS * FDIM;

    const int n0  = blockIdx.x * 128;
    const int m0  = blockIdx.y * 128;
    const int tid = threadIdx.x;
    const int wid = tid >> 5;
    const int lane = tid & 31;

    __shared__ __bf16 As[128][40];   // [row][k], k-pairs contiguous
    __shared__ __bf16 Bs[128][40];   // [n][k] (W transposed), k-pairs contiguous

    const int mr   = wid & 3;        // 32-row stripe (0..3)
    const int ncol = wid >> 2;       // 64-col half (0..1)

    v8f acc[2][4];
    #pragma unroll
    for (int i = 0; i < 2; ++i)
        #pragma unroll
        for (int j = 0; j < 4; ++j)
            #pragma unroll
            for (int e = 0; e < 8; ++e) acc[i][j][e] = 0.0f;

    for (int kk = 0; kk < 2048; kk += 32) {
        const float* X = (kk < 1024) ? x1 : x2;
        const int kb = kk & 1023;

        // ---- stage A tile: 128 rows x 32 k, convert f32 -> bf16 ----
        {
            int r = tid >> 1;
            int c = (tid & 1) * 16;
            const float4* src = (const float4*)(X + (size_t)(m0 + r) * FDIM + kb + c);
            #pragma unroll
            for (int q = 0; q < 4; ++q) {
                float4 f = src[q];
                int cc = c + q * 4;
                As[r][cc + 0] = tobf(f.x); As[r][cc + 1] = tobf(f.y);
                As[r][cc + 2] = tobf(f.z); As[r][cc + 3] = tobf(f.w);
            }
        }
        // ---- stage B tile transposed: Bs[n][k] = W[kk+k][n0+n] ----
        {
            int kl = tid >> 3;
            int nb = (tid & 7) * 16;
            const float4* src = (const float4*)(W + (size_t)(kk + kl) * FDIM + n0 + nb);
            #pragma unroll
            for (int q = 0; q < 4; ++q) {
                float4 f = src[q];
                int n = nb + q * 4;
                Bs[n + 0][kl] = tobf(f.x); Bs[n + 1][kl] = tobf(f.y);
                Bs[n + 2][kl] = tobf(f.z); Bs[n + 3][kl] = tobf(f.w);
            }
        }
        // prefetch next k-step source lines into cache (global_prefetch_b8)
        if (kk + 32 < 2048) {
            const float* Xn = ((kk + 32) < 1024) ? x1 : x2;
            __builtin_prefetch(Xn + (size_t)(m0 + (tid >> 1)) * FDIM + ((kk + 32) & 1023), 0, 0);
            __builtin_prefetch(W + (size_t)(kk + 32 + (tid >> 3)) * FDIM + n0 + (tid & 7) * 16, 0, 0);
        }
        __syncthreads();

        // ---- fragments (ISA 16-bit A 16x32 / B 32x16 layouts) ----
        v16bf af[2], bfv[4];
        const int am = lane & 15;
        const int k8 = (lane >> 4) << 3;
        #pragma unroll
        for (int i = 0; i < 8; ++i) {
            int k = 2 * i + ((i >= 4) ? 8 : 0) + k8;
            #pragma unroll
            for (int ti = 0; ti < 2; ++ti) {
                int row = mr * 32 + ti * 16 + am;
                af[ti][2 * i]     = As[row][k];
                af[ti][2 * i + 1] = As[row][k + 1];
            }
        }
        const int bn  = lane & 15;
        const int k16 = (lane >> 4) << 3;
        #pragma unroll
        for (int i = 0; i < 8; ++i) {
            int k = 2 * i + ((i >= 4) ? 8 : 0) + k16;
            #pragma unroll
            for (int tj = 0; tj < 4; ++tj) {
                int col = ncol * 64 + tj * 16 + bn;
                bfv[tj][2 * i]     = Bs[col][k];
                bfv[tj][2 * i + 1] = Bs[col][k + 1];
            }
        }
        #pragma unroll
        for (int ti = 0; ti < 2; ++ti)
            #pragma unroll
            for (int tj = 0; tj < 4; ++tj)
                acc[ti][tj] = __builtin_amdgcn_wmma_f32_16x16x32_bf16(
                    false, af[ti], false, bfv[tj], (short)0, acc[ti][tj], false, false);
        __syncthreads();
    }

    // ---- epilogue: bias + hardware tanh, store bf16 feature ----
    const int n_in = lane & 15;
    const int m8   = (lane >> 4) * 8;
    #pragma unroll
    for (int ti = 0; ti < 2; ++ti) {
        #pragma unroll
        for (int tj = 0; tj < 4; ++tj) {
            int col = n0 + ncol * 64 + tj * 16 + n_in;
            float bb = bias[col];
            int rowb = m0 + mr * 32 + ti * 16 + m8;
            #pragma unroll
            for (int i = 0; i < 8; ++i) {
                float val = tanh_fast(acc[ti][tj][i] + bb);
                feat[(size_t)(rowb + i) * FDIM + col] = f2bf_bits(val);
            }
        }
    }
}

// ===========================================================================
// Kernel 2: attention-gate.  grid = (S/16, B, 3), block = 256 (8 waves)
//   phase A: scores(16x512) = q @ feat^T      -> softmax -> P (bf16, LDS)
//   phase B: out(16x1024)   = (P @ q) * feat  -> d_out (concat offset br*F)
// ===========================================================================
#define QA_OFF 0            // __bf16 [16][40]   (1280 B, padded to 2048)
#define FT_OFF 2048         // __bf16 [512][40]  (40960 B)   phase-A only
#define SC_OFF 2048         // float  [16][520]  (33280 B)   overlaps FT (phase-ordered)
#define P_OFF  35328        // __bf16 [16][520]  (16640 B)
#define QT_OFF 51968        // __bf16 [128][40]  (10240 B)
#define SMEM_BYTES 62208

__global__ __launch_bounds__(256)
void attn_kernel(const float* __restrict__ v,
                 const float* __restrict__ t,
                 const float* __restrict__ a,
                 const unsigned short* __restrict__ featws,
                 float* __restrict__ out)
{
    const int br = blockIdx.z;
    const int b  = blockIdx.y;
    const int q0 = blockIdx.x * 16;
    const float* q = (br == 0) ? a : (br == 1) ? v : t;   // Oatv/Ovta/Otav
    const unsigned short* feat = featws + ((size_t)br * ROWS + (size_t)b * SDIM) * FDIM;
    const float* qb = q + (size_t)b * SDIM * FDIM;

    const int tid = threadIdx.x, wid = tid >> 5, lane = tid & 31;

    __shared__ __align__(16) char smem[SMEM_BYTES];
    __bf16 (*qA)[40] = (__bf16(*)[40])(smem + QA_OFF);
    __bf16 (*ft)[40] = (__bf16(*)[40])(smem + FT_OFF);
    float*  sc       = (float*)(smem + SC_OFF);
    __bf16* P        = (__bf16*)(smem + P_OFF);
    __bf16 (*qT)[40] = (__bf16(*)[40])(smem + QT_OFF);

    const unsigned long long feat_base = (unsigned long long)(const void*)feat;

    // ----------------- phase A: scores = q @ feat^T -----------------
    v8f accS[4];
    #pragma unroll
    for (int j = 0; j < 4; ++j)
        #pragma unroll
        for (int e = 0; e < 8; ++e) accS[j][e] = 0.0f;

    for (int f0 = 0; f0 < FDIM; f0 += 32) {
        if (tid < 128) {  // stage q A-tile 16x32, f32->bf16
            int r = tid >> 3, c = (tid & 7) * 4;
            float4 fq = *(const float4*)(qb + (size_t)(q0 + r) * FDIM + f0 + c);
            qA[r][c] = tobf(fq.x); qA[r][c + 1] = tobf(fq.y);
            qA[r][c + 2] = tobf(fq.z); qA[r][c + 3] = tobf(fq.w);
        }
        // stage feat tile (512 keys x 32 bf16) with async LDS DMA:
        //   global_load_async_to_lds_b128 (GVS: vdst=LDS addr, vaddr=off32, saddr)
        #pragma unroll
        for (int j = 0; j < 8; ++j) {
            int lin = tid * 8 + j;          // 2048 uint4 units
            int key = lin >> 2;
            int qd  = lin & 3;
            unsigned goff = (unsigned)key * (FDIM * 2u) + (unsigned)f0 * 2u + (unsigned)qd * 16u;
            unsigned loff = (unsigned)FT_OFF + (unsigned)key * 80u + (unsigned)qd * 16u;
            asm volatile("global_load_async_to_lds_b128 %0, %1, %2"
                         :
                         : "v"(loff), "v"(goff), "s"(feat_base)
                         : "memory");
        }
#if __has_builtin(__builtin_amdgcn_s_wait_asynccnt)
        __builtin_amdgcn_s_wait_asynccnt(0);
#else
        asm volatile("s_wait_asynccnt 0x0" ::: "memory");
#endif
        __syncthreads();

        v16bf af;
        int am = lane & 15, k8 = (lane >> 4) << 3;
        #pragma unroll
        for (int i = 0; i < 8; ++i) {
            int k = 2 * i + ((i >= 4) ? 8 : 0) + k8;
            af[2 * i] = qA[am][k]; af[2 * i + 1] = qA[am][k + 1];
        }
        int bn = lane & 15, k16 = (lane >> 4) << 3;
        #pragma unroll
        for (int tj = 0; tj < 4; ++tj) {
            v16bf bfr;
            int key = wid * 64 + tj * 16 + bn;
            #pragma unroll
            for (int i = 0; i < 8; ++i) {
                int k = 2 * i + ((i >= 4) ? 8 : 0) + k16;
                bfr[2 * i] = ft[key][k]; bfr[2 * i + 1] = ft[key][k + 1];
            }
            accS[tj] = __builtin_amdgcn_wmma_f32_16x16x32_bf16(
                false, af, false, bfr, (short)0, accS[tj], false, false);
        }
        __syncthreads();
    }

    // dump scores to LDS (reuses ft region; all waves past last read)
    {
        int n_in = lane & 15, m8 = (lane >> 4) * 8;
        #pragma unroll
        for (int tj = 0; tj < 4; ++tj)
            #pragma unroll
            for (int i = 0; i < 8; ++i)
                sc[(m8 + i) * 520 + wid * 64 + tj * 16 + n_in] = accS[tj][i];
    }
    __syncthreads();

    // ----------------- softmax (16 threads per row, half-wave shfl) -----------------
    {
        int row = tid >> 4, sub = tid & 15;
        float mx = -3.0e38f;
        for (int j = sub; j < SDIM; j += 16) mx = fmaxf(mx, sc[row * 520 + j]);
        #pragma unroll
        for (int off = 8; off; off >>= 1) mx = fmaxf(mx, __shfl_xor(mx, off, 16));
        float sum = 0.0f;
        for (int j = sub; j < SDIM; j += 16) {
            float e = __expf(sc[row * 520 + j] - mx);
            sc[row * 520 + j] = e;
            sum += e;
        }
        #pragma unroll
        for (int off = 8; off; off >>= 1) sum += __shfl_xor(sum, off, 16);
        float inv = 1.0f / sum;
        for (int j = sub; j < SDIM; j += 16)
            P[row * 520 + j] = tobf(sc[row * 520 + j] * inv);
    }
    __syncthreads();

    // ----------------- phase B: out = (P @ q) * feat -----------------
    for (int nc0 = 0; nc0 < FDIM; nc0 += 128) {
        v8f accO;
        #pragma unroll
        for (int e = 0; e < 8; ++e) accO[e] = 0.0f;

        for (int key0 = 0; key0 < SDIM; key0 += 32) {
            // stage qT[fl][kl] = q[key0+kl][nc0+fl] (transpose, f32->bf16)
            {
                int kl = tid >> 3;
                int fb = (tid & 7) * 16;
                const float4* src = (const float4*)(qb + (size_t)(key0 + kl) * FDIM + nc0 + fb);
                #pragma unroll
                for (int q4 = 0; q4 < 4; ++q4) {
                    float4 f = src[q4];
                    int fl = fb + q4 * 4;
                    qT[fl + 0][kl] = tobf(f.x); qT[fl + 1][kl] = tobf(f.y);
                    qT[fl + 2][kl] = tobf(f.z); qT[fl + 3][kl] = tobf(f.w);
                }
            }
            __syncthreads();

            v16bf af, bfr;
            int am = lane & 15, k8 = (lane >> 4) << 3;
            #pragma unroll
            for (int i = 0; i < 8; ++i) {
                int k = 2 * i + ((i >= 4) ? 8 : 0) + k8;
                af[2 * i]     = P[am * 520 + key0 + k];
                af[2 * i + 1] = P[am * 520 + key0 + k + 1];
            }
            int bn = lane & 15, k16 = (lane >> 4) << 3;
            int fl = wid * 16 + bn;
            #pragma unroll
            for (int i = 0; i < 8; ++i) {
                int k = 2 * i + ((i >= 4) ? 8 : 0) + k16;
                bfr[2 * i] = qT[fl][k]; bfr[2 * i + 1] = qT[fl][k + 1];
            }
            accO = __builtin_amdgcn_wmma_f32_16x16x32_bf16(
                false, af, false, bfr, (short)0, accO, false, false);
            __syncthreads();
        }

        // gate with feat, store to concatenated output
        int n_in = lane & 15, m8 = (lane >> 4) * 8;
        int fcol = nc0 + wid * 16 + n_in;
        #pragma unroll
        for (int i = 0; i < 8; ++i) {
            int rowq = q0 + m8 + i;
            float g = bf2f(feat[(size_t)rowq * FDIM + fcol]);
            out[((size_t)b * SDIM + rowq) * (3 * FDIM) + (size_t)br * FDIM + fcol] = accO[i] * g;
        }
    }
}

// ===========================================================================
extern "C" void kernel_launch(void* const* d_in, const int* in_sizes, int n_in,
                              void* d_out, int out_size, void* d_ws, size_t ws_size,
                              hipStream_t stream)
{
    const float* v   = (const float*)d_in[0];
    const float* t   = (const float*)d_in[1];
    const float* a   = (const float*)d_in[2];
    const float* Wtv = (const float*)d_in[3];
    const float* btv = (const float*)d_in[4];
    const float* Wta = (const float*)d_in[5];
    const float* bta = (const float*)d_in[6];
    const float* Wav = (const float*)d_in[7];
    const float* bav = (const float*)d_in[8];
    unsigned short* featws = (unsigned short*)d_ws;   // 48 MB bf16 features
    float* out = (float*)d_out;

    dim3 blk(256);
    dim3 g1(FDIM / 128, ROWS / 128, 3);
    branch_feat_kernel<<<g1, blk, 0, stream>>>(v, t, a, Wtv, btv, Wta, bta, Wav, bav, featws);

    dim3 g2(SDIM / 16, BDIM, 3);
    attn_kernel<<<g2, blk, 0, stream>>>(v, t, a, featws, out);
}
